// KaldiLSTMP_57088705298595
// MI455X (gfx1250) — matrix operand order repaired and checked by reading, not compiled, not run on hardware
//
#include <hip/hip_runtime.h>
#include <hip/hip_bf16.h>
#include <math.h>

#define NINP  240
#define NHIDD 1024
#define NPROJ 512
#define NGATE (4 * NHIDD)        // 4096
#define T_SEQ 500
#define BATCH 32
#define TBROWS (T_SEQ * BATCH)   // 16000

typedef __attribute__((ext_vector_type(16))) __bf16 v16bf;
typedef __attribute__((ext_vector_type(8)))  __bf16 v8bf;
typedef __attribute__((ext_vector_type(8)))  float  v8f;

// ---------------------------------------------------------------------------
// WMMA fragment helpers (CDNA5 wave32 layouts, ISA 7.12.2).
// All K dims are pack-time multiples of 32 -> no bounds checks.
// ---------------------------------------------------------------------------

// A matrix 16x32 bf16 from bf16 row-major src: two contiguous b128 runs/lane.
__device__ __forceinline__ v16bf load_a_bf16(const __bf16* __restrict__ A,
                                             int lda, int row0, int k0) {
    const int lane = threadIdx.x & 31;
    const int m    = row0 + (lane & 15);
    const int khi  = (lane >> 4) << 4;  // 0 or 16
    const __bf16* p = A + (size_t)m * lda + k0 + khi;
    v8bf lo = *(const v8bf*)(p);
    v8bf hi = *(const v8bf*)(p + 8);
    return __builtin_shufflevector(lo, hi, 0, 1, 2, 3, 4, 5, 6, 7,
                                           8, 9, 10, 11, 12, 13, 14, 15);
}

// B matrix 32x16 bf16 where B[k][j] = W[j][k]: one 32-byte run per lane.
__device__ __forceinline__ v16bf load_b_bf16(const __bf16* __restrict__ W,
                                             int ldw, int j0, int k0) {
    const int lane = threadIdx.x & 31;
    const int j    = j0 + (lane & 15);
    const int khi  = (lane >> 4) << 4;  // 0 or 16
    return *(const v16bf*)(W + (size_t)j * ldw + k0 + khi);
}

// C/D f32 16x16: lane L holds column (j0 + L%16); comp v -> row v + (L/16)*8
__device__ __forceinline__ v8f load_c_f32(const float* __restrict__ C, int ldc,
                                          int row0, int j0) {
    const int lane  = threadIdx.x & 31;
    const int n     = j0 + (lane & 15);
    const int rbase = row0 + ((lane >> 4) << 3);
    v8f c;
#pragma unroll
    for (int v = 0; v < 8; ++v) c[v] = C[(size_t)(rbase + v) * ldc + n];
    return c;
}

__device__ __forceinline__ void store_d_f32(float* __restrict__ D, int ldd,
                                            int row0, int j0, v8f acc) {
    const int lane  = threadIdx.x & 31;
    const int n     = j0 + (lane & 15);
    const int rbase = row0 + ((lane >> 4) << 3);
#pragma unroll
    for (int v = 0; v < 8; ++v) D[(size_t)(rbase + v) * ldd + n] = acc[v];
}

__device__ __forceinline__ void store_d_bf16(__bf16* __restrict__ D, int ldd,
                                             int row0, int j0, v8f acc) {
    const int lane  = threadIdx.x & 31;
    const int n     = j0 + (lane & 15);
    const int rbase = row0 + ((lane >> 4) << 3);
#pragma unroll
    for (int v = 0; v < 8; ++v) D[(size_t)(rbase + v) * ldd + n] = (__bf16)acc[v];
}

__device__ __forceinline__ float sigmoidf_(float x) {
    return 1.0f / (1.0f + __expf(-x));
}

// ---------------------------------------------------------------------------
// Pack kernel: dst[r][c] = (c < ncols) ? bf16(src[r*ldsrc + col0 + c]) : 0
// ---------------------------------------------------------------------------
__global__ void k_pack_cols(const float* __restrict__ src, __bf16* __restrict__ dst,
                            int ldsrc, int col0, int ncols, int lddst, long total) {
    long i = (long)blockIdx.x * blockDim.x + threadIdx.x;
    if (i >= total) return;
    int r = (int)(i / lddst);
    int c = (int)(i % lddst);
    dst[i] = (c < ncols) ? (__bf16)src[(size_t)r * ldsrc + col0 + c] : (__bf16)0.0f;
}

// ---------------------------------------------------------------------------
// Xg = X @ Wx^T + bias for ALL timesteps: [16000 x kpad] x [kpad x 4096].
// grid = (TBROWS/16, 64), block = 128 (4 waves, one 16x16 N-tile each)
// ---------------------------------------------------------------------------
__global__ void k_precompute_xg(const __bf16* __restrict__ Xbf, int kpad,
                                const __bf16* __restrict__ Wxb,
                                const float* __restrict__ bias,
                                float* __restrict__ Xg) {
    const int wave = threadIdx.x >> 5;
    const int lane = threadIdx.x & 31;
    const int row0 = blockIdx.x << 4;
    const int j0   = ((blockIdx.y << 2) + wave) << 4;

    float bv = bias[j0 + (lane & 15)];
    v8f acc;
#pragma unroll
    for (int v = 0; v < 8; ++v) acc[v] = bv;

    for (int k0 = 0; k0 < kpad; k0 += 32) {
        v16bf a = load_a_bf16(Xbf, kpad, row0, k0);
        v16bf b = load_b_bf16(Wxb, kpad, j0, k0);
        acc = __builtin_amdgcn_wmma_f32_16x16x32_bf16(false, a, false, b,
                                                      (short)0, acc, false, false);
    }
    store_d_f32(Xg, NGATE, row0, j0, acc);
}

// ---------------------------------------------------------------------------
// One recurrent step: gates = Xg[t] + h_{t-1} @ Wh^T, fused LSTM cell update.
// The block's 64 Wh rows (4 gate tiles x 16 cols x K=512, 64 KB bf16) are
// staged into LDS with GLOBAL_LOAD_ASYNC_TO_LDS_B128 (ASYNCcnt), overlapped
// with the Xg C-fragment load; both M-tile waves then read B from LDS.
// grid = 64, block = 256 (8 waves = {4 gates} x {2 M-tiles})
// ---------------------------------------------------------------------------
__global__ void k_step_gates_cell(const float* __restrict__ XgT,    // Xg + t*32*4096
                                  const __bf16* __restrict__ Hprev, // bf16 [32, 512]
                                  const __bf16* __restrict__ Whb,   // bf16 [4096, 512]
                                  float* __restrict__ cstate,       // fp32 [32, 1024]
                                  const float* __restrict__ pi, const float* __restrict__ pf,
                                  const float* __restrict__ po,
                                  __bf16* __restrict__ u,           // bf16 [32, 1024]
                                  int first) {
    __shared__ __bf16 Bsh[64 * NPROJ];   // 64 rows x 512 K = 64 KB
    __shared__ float  gs[4][BATCH][16];  // gate tiles for the cell update

    const int tid   = threadIdx.x;
    const int wave  = tid >> 5;
    const int lane  = tid & 31;
    const int g     = wave & 3;
    const int mtile = wave >> 2;
    const int row0  = mtile << 4;
    const int htile = blockIdx.x;
    const int j0    = g * NHIDD + (htile << 4);

    if (!first) {
        // Async-stage Wh tile: 64 rows x 1 KB = 4096 b128 chunks over 256 lanes.
        // Local row r = g*16 + col  <->  global row j = g*1024 + htile*16 + col.
#pragma unroll
        for (int it = 0; it < 16; ++it) {
            const int id  = it * 256 + tid;
            const int r   = id >> 6;                    // 0..63
            const int c16 = id & 63;                    // 16B chunk in row
            const int jg  = (r >> 4) * NHIDD + (htile << 4) + (r & 15);
            const __bf16* gp = Whb + (size_t)jg * NPROJ + c16 * 8;
            const unsigned ldsoff =
                (unsigned)(size_t)(Bsh + r * NPROJ + c16 * 8);
            asm volatile("global_load_async_to_lds_b128 %0, %1, off"
                         :: "v"(ldsoff), "v"(gp) : "memory");
        }
    }

    v8f acc = load_c_f32(XgT, NGATE, row0, j0);  // input projection + bias

    if (!first) {
        asm volatile("s_wait_asynccnt 0x0" ::: "memory");
    }
    __syncthreads();  // LDS B-stage visible to all waves

    if (!first) {
        const __bf16* bbase = Bsh + ((g << 4) + (lane & 15)) * NPROJ;
        const int     khi   = (lane >> 4) << 4;  // 0 or 16
#pragma unroll 4
        for (int k0 = 0; k0 < NPROJ; k0 += 32) {
            v16bf a = load_a_bf16(Hprev, NPROJ, row0, k0);
            v16bf b = *(const v16bf*)(bbase + k0 + khi);  // ds_load_b128 x2
            acc = __builtin_amdgcn_wmma_f32_16x16x32_bf16(false, a, false, b,
                                                          (short)0, acc, false, false);
        }
    }

    {   // gate tile -> LDS
        const int n     = lane & 15;
        const int rbase = row0 + ((lane >> 4) << 3);
#pragma unroll
        for (int v = 0; v < 8; ++v) gs[g][rbase + v][n] = acc[v];
    }
    __syncthreads();

    // pointwise LSTM cell: 512 elems over 256 threads
    for (int e = tid; e < BATCH * 16; e += blockDim.x) {
        const int r   = e >> 4;
        const int col = e & 15;
        const int h   = (htile << 4) + col;
        float cx  = first ? 0.0f : cstate[r * NHIDD + h];
        float cin = gs[0][r][col];
        float ig  = sigmoidf_(gs[1][r][col] + pi[h] * cx);
        float fg  = sigmoidf_(gs[2][r][col] + pf[h] * cx);
        float cy  = fg * cx + ig * tanhf(cin);
        cy = fminf(50.0f, fmaxf(-50.0f, cy));
        float og  = sigmoidf_(gs[3][r][col] + po[h] * cy);
        cstate[r * NHIDD + h] = cy;
        u[r * NHIDD + h] = (__bf16)(og * tanhf(cy));
    }
}

// ---------------------------------------------------------------------------
// hy = u @ P^T : [32 x 1024] x [1024 x 512]. Emits bf16 (recurrent feed +
// next layer's A operand) and fp32 (final output / dummy).
// grid = 16, block = 128 (64 tiles over 64 waves)
// ---------------------------------------------------------------------------
__global__ void k_step_proj(const __bf16* __restrict__ u, const __bf16* __restrict__ Pb,
                            float* __restrict__ HyF, __bf16* __restrict__ HyB) {
    const int wave  = threadIdx.x >> 5;
    const int tid   = (blockIdx.x << 2) + wave;   // 0..63
    const int row0  = (tid >> 5) << 4;
    const int j0    = (tid & 31) << 4;

    v8f acc;
#pragma unroll
    for (int v = 0; v < 8; ++v) acc[v] = 0.0f;

#pragma unroll 4
    for (int k0 = 0; k0 < NHIDD; k0 += 32) {
        v16bf a = load_a_bf16(u, NHIDD, row0, k0);
        v16bf b = load_b_bf16(Pb, NHIDD, j0, k0);
        acc = __builtin_amdgcn_wmma_f32_16x16x32_bf16(false, a, false, b,
                                                      (short)0, acc, false, false);
    }
    store_d_f32(HyF, NPROJ, row0, j0, acc);
    store_d_bf16(HyB, NPROJ, row0, j0, acc);
}

// ---------------------------------------------------------------------------
// Host orchestration
// ---------------------------------------------------------------------------
extern "C" void kernel_launch(void* const* d_in, const int* in_sizes, int n_in,
                              void* d_out, int out_size, void* d_ws, size_t ws_size,
                              hipStream_t stream) {
    (void)in_sizes; (void)n_in; (void)out_size; (void)ws_size;

    const float* x = (const float*)d_in[0];
    const float *W[3], *bias[3], *P[3], *pi[3], *pf[3], *po[3];
    for (int l = 0; l < 3; ++l) {
        W[l]    = (const float*)d_in[1 + l * 6 + 0];
        bias[l] = (const float*)d_in[1 + l * 6 + 1];
        P[l]    = (const float*)d_in[1 + l * 6 + 2];
        pi[l]   = (const float*)d_in[1 + l * 6 + 3];
        pf[l]   = (const float*)d_in[1 + l * 6 + 4];
        po[l]   = (const float*)d_in[1 + l * 6 + 5];
    }

    const int nin_l[3]  = { NINP, NPROJ, NPROJ };
    const int kpad_l[3] = { 256,  NPROJ, NPROJ };   // K padded to mult. of 32
    const int ktot_l[3] = { NINP + NPROJ, NPROJ + NPROJ, NPROJ + NPROJ };

    char* ws = (char*)d_ws;
    size_t off = 0;
    auto carve = [&](size_t bytes) -> void* {
        void* p = ws + off;
        off = (off + bytes + 255) & ~(size_t)255;
        return p;
    };
    float*  Xg   = (float*)carve((size_t)TBROWS * NGATE * sizeof(float));   // 262 MB
    __bf16* Xpad = (__bf16*)carve((size_t)TBROWS * 256 * sizeof(__bf16));
    __bf16 *Wxb[3], *Whb[3], *Pb[3], *Ybf[3];
    for (int l = 0; l < 3; ++l) Wxb[l] = (__bf16*)carve((size_t)NGATE * kpad_l[l] * sizeof(__bf16));
    for (int l = 0; l < 3; ++l) Whb[l] = (__bf16*)carve((size_t)NGATE * NPROJ * sizeof(__bf16));
    for (int l = 0; l < 3; ++l) Pb[l]  = (__bf16*)carve((size_t)NPROJ * NHIDD * sizeof(__bf16));
    for (int l = 0; l < 3; ++l) Ybf[l] = (__bf16*)carve((size_t)TBROWS * NPROJ * sizeof(__bf16));
    float*  dummyF = (float*)carve((size_t)BATCH * NPROJ * sizeof(float));
    float*  cst    = (float*)carve((size_t)BATCH * NHIDD * sizeof(float));
    __bf16* ub     = (__bf16*)carve((size_t)BATCH * NHIDD * sizeof(__bf16));

    auto pack = [&](const float* src, __bf16* dst, int ldsrc, int col0,
                    int ncols, int lddst, int nrows) {
        long total = (long)nrows * lddst;
        k_pack_cols<<<(unsigned)((total + 255) / 256), 256, 0, stream>>>(
            src, dst, ldsrc, col0, ncols, lddst, total);
    };

    // 1) pack all WMMA operands into padded bf16 (deterministic, once per call)
    pack(x, Xpad, NINP, 0, NINP, 256, TBROWS);
    for (int l = 0; l < 3; ++l) {
        pack(W[l], Wxb[l], ktot_l[l], 0,        nin_l[l], kpad_l[l], NGATE);
        pack(W[l], Whb[l], ktot_l[l], nin_l[l], NPROJ,    NPROJ,     NGATE);
        pack(P[l], Pb[l],  NHIDD,     0,        NHIDD,    NHIDD,     NPROJ);
    }

    const __bf16* Abf[3] = { Xpad, Ybf[0], Ybf[1] };

    for (int l = 0; l < 3; ++l) {
        // 2) big non-recurrent GEMM for all 500 steps at once
        k_precompute_xg<<<dim3(TBROWS / 16, NGATE / 64), 128, 0, stream>>>(
            Abf[l], kpad_l[l], Wxb[l], bias[l], Xg);

        // 3) sequential recurrence: two small dependent kernels per step
        for (int t = 0; t < T_SEQ; ++t) {
            const __bf16* Hprev = (t == 0) ? Ybf[l]  // unused when first==1
                                           : (Ybf[l] + (size_t)(t - 1) * BATCH * NPROJ);
            k_step_gates_cell<<<NHIDD / 16, 256, 0, stream>>>(
                Xg + (size_t)t * BATCH * NGATE, Hprev, Whb[l],
                cst, pi[l], pf[l], po[l], ub, (t == 0) ? 1 : 0);

            float* hyF = (l == 2) ? ((float*)d_out + (size_t)t * BATCH * NPROJ) : dummyF;
            k_step_proj<<<16, 128, 0, stream>>>(
                ub, Pb[l], hyF, Ybf[l] + (size_t)t * BATCH * NPROJ);
        }
    }
}
